// ARMA_41360535061064
// MI455X (gfx1250) — matrix operand order repaired and checked by reading, not compiled, not run on hardware
//
#include <hip/hip_runtime.h>
#include <hip/hip_bf16.h>
#include <stdint.h>

// ---------------------------------------------------------------------------
// ARMA GNN forward for MI455X (gfx1250, wave32, WMMA).
// Dense + grouped GEMMs run on v_wmma_f32_16x16x32_bf16 (bf16 A/B, f32 acc).
// Weight panels are staged into LDS by the Tensor Data Mover (one
// tensor_load_to_lds per block, hardware row padding -> conflict-free
// ds_load_b128 fragment reads), with a cooperative-copy fallback if the TDM
// builtin is unavailable. Laplacian scatter uses float4 loads + global f32
// atomics (tgt sorted -> L2-friendly).
// ---------------------------------------------------------------------------

#define NNODES 50000
#define NEDGES 800000
#define INF    128
#define OUTF   64
#define HF     32
#define WIDTH  3

typedef __bf16 v16bf __attribute__((ext_vector_type(16)));
typedef __bf16 v8bf  __attribute__((ext_vector_type(8)));
typedef float  v8f   __attribute__((ext_vector_type(8)));

__device__ __forceinline__ __bf16 f2bf(float f) {
    union { float f; unsigned u; } v; v.f = f;
    unsigned r = v.u + 0x7FFFu + ((v.u >> 16) & 1u);   // round-to-nearest-even
    union { unsigned short u; __bf16 b; } o;
    o.u = (unsigned short)(r >> 16);
    return o.b;
}

// ---------------------------------------------------------------------------
// Tensor Data Mover panel staging (gfx1250).
// Loads a [nrows x rowElems] bf16 panel (row-major, contiguous) into LDS,
// inserting 4 DWORDs (16 B) of padding after every row via the TDM pad unit.
//   padCode: pad_interval code; row bytes = (1<<padCode)*8  (5 -> 256B, 3 -> 64B)
// D# layout per CDNA5 ISA sec 8.3/8.4 (group0: count/lds/global/type,
// group1: data_size|pad cfg, tensor dims, tile dims, stride).
// ---------------------------------------------------------------------------
#if __has_builtin(__builtin_amdgcn_tensor_load_to_lds)
#define USE_TDM 1
#if __has_include(<hip/amd_detail/amd_gfx1250_TDM.h>)
#define TDM_SIX_ARGS 1
#endif
#endif

#if defined(USE_TDM)
typedef unsigned int u32x4 __attribute__((ext_vector_type(4)));
typedef int          i32x4 __attribute__((ext_vector_type(4)));
typedef int          i32x8 __attribute__((ext_vector_type(8)));

__device__ __forceinline__ void tdm_load_panel(const __bf16* gsrc, unsigned lds_off,
                                               int rowElems, int nrows, int padCode) {
    unsigned long long ga = (unsigned long long)(size_t)gsrc;
    u32x4 g0 = { 1u,                                   // count=1, is_restore=0
                 lds_off,                              // lds_addr (bytes)
                 (unsigned)ga,                         // global_addr[31:0]
                 (unsigned)((ga >> 32) & 0x1FFFFFFu)   // global_addr[56:32]
                     | (2u << 30) };                   // type=2 ("image")
    i32x8 g1 = { (int)((1u << 16)                      // data_size=1 -> 2 bytes
                     | (1u << 20)                      // pad_enable
                     | ((unsigned)padCode << 22)       // pad_interval
                     | (3u << 25)),                    // pad_amount = 4 DWORDs (16B)
                 (int)((rowElems & 0xFFFF) << 16),     // tensor_dim0[15:0]
                 (int)(((rowElems >> 16) & 0xFFFF)     // tensor_dim0[31:16]
                     | ((nrows & 0xFFFF) << 16)),      // tensor_dim1[15:0]
                 (int)(((nrows >> 16) & 0xFFFF)        // tensor_dim1[31:16]
                     | ((rowElems & 0xFFFF) << 16)),   // tile_dim0
                 (int)(nrows & 0xFFFF),                // tile_dim1 (tile_dim2=0)
                 rowElems,                             // tensor_dim0_stride[31:0]
                 0, 0 };
    i32x4 gz = { 0, 0, 0, 0 };
#if defined(TDM_SIX_ARGS)
    i32x8 gz8 = { 0, 0, 0, 0, 0, 0, 0, 0 };
    __builtin_amdgcn_tensor_load_to_lds(g0, g1, gz, gz, gz8, 0);
#else
    __builtin_amdgcn_tensor_load_to_lds(g0, g1, gz, gz, 0);
#endif
}
#endif

// ---------------------------------------------------------------------------
// Elementwise / prep kernels
// ---------------------------------------------------------------------------
__global__ __launch_bounds__(256) void cast_bf16_kernel(
    const float* __restrict__ in, __bf16* __restrict__ out, int n) {
    int i = blockIdx.x * blockDim.x + threadIdx.x;
    if (i < n) out[i] = f2bf(in[i]);
}

// W [K,C] f32 -> Wt [C,K] bf16
__global__ __launch_bounds__(256) void transpose_cast_kernel(
    const float* __restrict__ W, __bf16* __restrict__ Wt, int K, int C) {
    int i = blockIdx.x * blockDim.x + threadIdx.x;
    if (i >= K * C) return;
    int k = i / C, c = i % C;
    Wt[(size_t)c * K + k] = f2bf(W[(size_t)k * C + c]);
}

// W [G,HI,HO] f32 -> Wt [G,HO,HI] bf16
__global__ __launch_bounds__(256) void transpose_cast_grouped_kernel(
    const float* __restrict__ W, __bf16* __restrict__ Wt, int G, int HI, int HO) {
    int i = blockIdx.x * blockDim.x + threadIdx.x;
    if (i >= G * HI * HO) return;
    int g = i / (HI * HO);
    int r = i % (HI * HO);
    int h = r / HO, o = r % HO;
    Wt[((size_t)g * HO + o) * HI + h] = f2bf(W[((size_t)g * HI + h) * HO + o]);
}

__global__ __launch_bounds__(256) void degree_kernel(
    const int* __restrict__ tgt, float* __restrict__ deg, int E) {
    int i = blockIdx.x * blockDim.x + threadIdx.x;
    if (i < E) unsafeAtomicAdd(&deg[tgt[i]], 1.0f);
}

__global__ __launch_bounds__(256) void invert_kernel(float* __restrict__ d, int n) {
    int i = blockIdx.x * blockDim.x + threadIdx.x;
    if (i < n) d[i] = (d[i] > 0.0f) ? (1.0f / d[i]) : 0.0f;
}

// A = relu(L + S), stored bf16 (feeds next grouped WMMA)
__global__ __launch_bounds__(256) void add_relu_bf16_kernel(
    const float* __restrict__ L, const float* __restrict__ S,
    __bf16* __restrict__ A, int n) {
    int i = blockIdx.x * blockDim.x + threadIdx.x;
    if (i < n) {
        float v = L[i] + S[i];
        A[i] = f2bf(v > 0.0f ? v : 0.0f);
    }
}

// out[n,o] = mean_w relu(L[n,3o+w] + S[n,3o+w])
__global__ __launch_bounds__(256) void final_mean_kernel(
    const float* __restrict__ L, const float* __restrict__ S,
    float* __restrict__ out, int total) {   // total = N*OUTF
    int i = blockIdx.x * blockDim.x + threadIdx.x;
    if (i >= total) return;
    int node = i / OUTF, c = i % OUTF;
    size_t base = (size_t)node * (OUTF * WIDTH) + (size_t)c * WIDTH;
    float acc = 0.0f;
#pragma unroll
    for (int w = 0; w < WIDTH; ++w) {
        float v = L[base + w] + S[base + w];
        acc += (v > 0.0f ? v : 0.0f);
    }
    out[i] = acc * (1.0f / WIDTH);
}

// ---------------------------------------------------------------------------
// Normalized-Laplacian edge scatter: Out[t] += (X[t]-X[s]) * invdeg[t]
// one thread per (edge, 4-feature chunk)
// ---------------------------------------------------------------------------
template <int F>
__global__ __launch_bounds__(256) void laplacian_kernel(
    const float* __restrict__ X, const int* __restrict__ src,
    const int* __restrict__ tgt, const float* __restrict__ invdeg,
    float* __restrict__ Out, int E) {
    const int CH = F / 4;
    int tid = blockIdx.x * blockDim.x + threadIdx.x;
    int e = tid / CH;
    if (e >= E) return;
    int c = (tid % CH) * 4;
    int s = src[e], t = tgt[e];
    float inv = invdeg[t];
    const float4 xt = *reinterpret_cast<const float4*>(X + (size_t)t * F + c);
    const float4 xs = *reinterpret_cast<const float4*>(X + (size_t)s * F + c);
    float* o = Out + (size_t)t * F + c;
    unsafeAtomicAdd(o + 0, (xt.x - xs.x) * inv);
    unsafeAtomicAdd(o + 1, (xt.y - xs.y) * inv);
    unsafeAtomicAdd(o + 2, (xt.z - xs.z) * inv);
    unsafeAtomicAdd(o + 3, (xt.w - xs.w) * inv);
}

// ---------------------------------------------------------------------------
// Dense GEMM: Y[N,COUT] = Xbf[N,128] @ Wt[COUT,128]^T + bias
// 256 threads = 8 waves; weight panel staged into LDS by TDM (row stride
// 136 bf16 = 272 B -> bank-conflict-free b128 reads). One wave per 16-row
// tile. Per column tile, all four K-step B fragments are issued before the
// four WMMAs so the LDS waits are staged instead of per-WMMA.
// A layout (ISA 16-bit A 16x32): lane&15 = row; lane>>4 selects K+0/K+8;
//   elements 0..7 -> K..K+7, elements 8..15 -> K+16..K+23.
// B layout: lane&15 = col; lane>>4 selects K+0/K+8; elements in the same K order as A.
// ---------------------------------------------------------------------------
#define BSTRIDE 136   // 128 + 8 bf16 pad (matches TDM pad: 256B row + 16B)

template <int COUT>
__global__ __launch_bounds__(256) void gemm_bf16_kernel(
    const __bf16* __restrict__ X, const __bf16* __restrict__ Wt,
    const float* __restrict__ bias, float* __restrict__ Y, int ntiles) {
    __shared__ __align__(16) __bf16 Blds[COUT * BSTRIDE];

#if defined(USE_TDM)
    if (threadIdx.x < 32) {          // one wave issues the DMA
        tdm_load_panel(Wt, (unsigned)(size_t)&Blds[0], INF, COUT, /*padCode=*/5);
        __builtin_amdgcn_s_wait_tensorcnt(0);
    }
    __syncthreads();
#else
    {
        const uint4* wsrc = reinterpret_cast<const uint4*>(Wt);
        uint4*       wdst = reinterpret_cast<uint4*>(Blds);
        for (int i = threadIdx.x; i < COUT * 16; i += 256)
            wdst[(i >> 4) * 17 + (i & 15)] = wsrc[i];
    }
    __syncthreads();
#endif

    const int lane = threadIdx.x & 31;
    const int wave = threadIdx.x >> 5;
    const int tile = blockIdx.x * 8 + wave;
    if (tile >= ntiles) return;                    // wave-uniform: EXEC stays full
    const int m0  = tile * 16;
    const int row = m0 + (lane & 15);
    const int hi  = lane >> 4;

    v16bf a[4];
#pragma unroll
    for (int s = 0; s < 4; ++s) {
        const int k0 = 32 * s + hi * 8;
        const v8bf lo  = *reinterpret_cast<const v8bf*>(X + (size_t)row * INF + k0);
        const v8bf hi8 = *reinterpret_cast<const v8bf*>(X + (size_t)row * INF + k0 + 16);
#pragma unroll
        for (int i = 0; i < 8; ++i) { a[s][i] = lo[i]; a[s][i + 8] = hi8[i]; }
    }

#pragma unroll
    for (int j = 0; j < COUT / 16; ++j) {
        const int col = j * 16 + (lane & 15);
        v16bf b[4];
#pragma unroll
        for (int s = 0; s < 4; ++s) {              // issue all 8 ds_loads first
            const int kb = 32 * s + hi * 8;
            const v8bf blo = *reinterpret_cast<const v8bf*>(Blds + col * BSTRIDE + kb);
            const v8bf bhi = *reinterpret_cast<const v8bf*>(Blds + col * BSTRIDE + kb + 16);
#pragma unroll
            for (int i = 0; i < 8; ++i) { b[s][i] = blo[i]; b[s][i + 8] = bhi[i]; }
        }
        v8f c = {};
#pragma unroll
        for (int s = 0; s < 4; ++s)
            c = __builtin_amdgcn_wmma_f32_16x16x32_bf16(
                    false, a[s], false, b[s], (short)0, c, false, false);
        const float bv = bias[col];
#pragma unroll
        for (int r = 0; r < 8; ++r)
            Y[(size_t)(m0 + r + 8 * hi) * COUT + col] = c[r] + bv;
    }
}

// ---------------------------------------------------------------------------
// Grouped GEMM (Conv1d k=1, groups=3): per group g,
//   Y[:, g*HO+o] = A[:, g*32+h] @ Wt[g,o,h] + bias[g*HO+o]
// K = 32 exactly -> one WMMA per 16x16 tile. Weight panel staged by TDM with
// 64B rows + 16B pad (stride 40 bf16) -> conflict-free b128 fragment reads.
// ---------------------------------------------------------------------------
#define GSTRIDE 40    // 32 + 8 bf16 pad

template <int HO>
__global__ __launch_bounds__(256) void grouped_gemm_kernel(
    const __bf16* __restrict__ A,   // [N, 96] bf16
    const __bf16* __restrict__ Wt,  // [3, HO, 32] bf16
    const float* __restrict__ bias, // [3*HO]
    float* __restrict__ Y, int ntiles) {
    __shared__ __align__(16) __bf16 Wlds[WIDTH * HO * GSTRIDE];

#if defined(USE_TDM)
    if (threadIdx.x < 32) {
        tdm_load_panel(Wt, (unsigned)(size_t)&Wlds[0], HF, WIDTH * HO, /*padCode=*/3);
        __builtin_amdgcn_s_wait_tensorcnt(0);
    }
    __syncthreads();
#else
    {
        const uint4* wsrc = reinterpret_cast<const uint4*>(Wt);
        uint4*       wdst = reinterpret_cast<uint4*>(Wlds);
        for (int i = threadIdx.x; i < WIDTH * HO * 4; i += 256)
            wdst[(i >> 2) * 5 + (i & 3)] = wsrc[i];
    }
    __syncthreads();
#endif

    const int lane = threadIdx.x & 31;
    const int wave = threadIdx.x >> 5;
    const int tile = blockIdx.x * 8 + wave;
    if (tile >= ntiles) return;
    const int m0  = tile * 16;
    const int row = m0 + (lane & 15);
    const int hi  = lane >> 4;
    const int CO  = WIDTH * HO;

#pragma unroll
    for (int g = 0; g < WIDTH; ++g) {
        const size_t abase = (size_t)row * (WIDTH * HF) + g * HF + hi * 8;
        const v8bf lo  = *reinterpret_cast<const v8bf*>(A + abase);
        const v8bf hi8 = *reinterpret_cast<const v8bf*>(A + abase + 16);
        v16bf af;
#pragma unroll
        for (int i = 0; i < 8; ++i) { af[i] = lo[i]; af[i + 8] = hi8[i]; }

        v16bf b[HO / 16];
#pragma unroll
        for (int t = 0; t < HO / 16; ++t) {        // issue all ds_loads first
            const int col = t * 16 + (lane & 15);
            const __bf16* wp = Wlds + (g * HO + col) * GSTRIDE + hi * 8;
            const v8bf blo = *reinterpret_cast<const v8bf*>(wp);
            const v8bf bhi = *reinterpret_cast<const v8bf*>(wp + 16);
#pragma unroll
            for (int i = 0; i < 8; ++i) { b[t][i] = blo[i]; b[t][i + 8] = bhi[i]; }
        }
#pragma unroll
        for (int t = 0; t < HO / 16; ++t) {
            const int col = t * 16 + (lane & 15);
            v8f c = {};
            c = __builtin_amdgcn_wmma_f32_16x16x32_bf16(
                    false, af, false, b[t], (short)0, c, false, false);
            const float bv = bias[g * HO + col];
#pragma unroll
            for (int r = 0; r < 8; ++r)
                Y[(size_t)(m0 + r + 8 * hi) * CO + g * HO + col] = c[r] + bv;
        }
    }
}

// ---------------------------------------------------------------------------
// Host driver
// ---------------------------------------------------------------------------
extern "C" void kernel_launch(void* const* d_in, const int* in_sizes, int n_in,
                              void* d_out, int out_size, void* d_ws, size_t ws_size,
                              hipStream_t stream) {
    const float* data   = (const float*)d_in[0];
    const int*   src    = (const int*)  d_in[1];
    const int*   tgt    = (const int*)  d_in[2];
    const float* W_pre  = (const float*)d_in[3];
    const float* b_pre  = (const float*)d_in[4];
    const float* T_pre  = (const float*)d_in[5];
    const float* bT_pre = (const float*)d_in[6];
    const float* W_blk  = (const float*)d_in[7];
    const float* b_blk  = (const float*)d_in[8];
    const float* T_blk  = (const float*)d_in[9];
    const float* bT_blk = (const float*)d_in[10];
    const float* W_post = (const float*)d_in[11];
    const float* b_post = (const float*)d_in[12];
    const float* T_post = (const float*)d_in[13];
    const float* bT_post= (const float*)d_in[14];

    const int N = NNODES, E = NEDGES;
    const int HW = HF * WIDTH;          // 96
    const int OW = OUTF * WIDTH;        // 192

    uint8_t* p = (uint8_t*)d_ws;
    auto carve = [&](size_t bytes) -> void* {
        void* r = (void*)p;
        p += (bytes + 255) & ~(size_t)255;
        return r;
    };
    __bf16* Xbf     = (__bf16*)carve((size_t)N * INF * 2);
    __bf16* Wpre_t  = (__bf16*)carve((size_t)HW * INF * 2);
    __bf16* Tpre_t  = (__bf16*)carve((size_t)HW * INF * 2);
    __bf16* Tblk_t  = (__bf16*)carve((size_t)HW * INF * 2);
    __bf16* Tpost_t = (__bf16*)carve((size_t)OW * INF * 2);
    __bf16* Wblk_t  = (__bf16*)carve((size_t)WIDTH * HF * HF * 2);
    __bf16* Wpost_t = (__bf16*)carve((size_t)WIDTH * HF * OUTF * 2);
    float*  invdeg  = (float*) carve((size_t)N * 4);
    float*  G       = (float*) carve((size_t)N * OW * 4);   // GEMM outputs
    float*  L       = (float*) carve((size_t)N * OW * 4);   // Laplacian accum
    float*  S       = (float*) carve((size_t)N * OW * 4);   // skip branch
    __bf16* Abf     = (__bf16*)carve((size_t)N * HW * 2);   // activations (bf16)

    const int TPB = 256;
    auto blocks = [](long long n, int t) { return (int)((n + t - 1) / t); };
    const int ntiles   = N / 16;                 // 3125 exact
    const int gemm_grd = (ntiles + 7) / 8;       // 8 waves / block

    // --- degree -> invdeg ---
    hipMemsetAsync(invdeg, 0, (size_t)N * 4, stream);
    degree_kernel<<<blocks(E, TPB), TPB, 0, stream>>>(tgt, invdeg, E);
    invert_kernel<<<blocks(N, TPB), TPB, 0, stream>>>(invdeg, N);

    // --- casts / weight transposes ---
    cast_bf16_kernel<<<blocks((long long)N * INF, TPB), TPB, 0, stream>>>(data, Xbf, N * INF);
    transpose_cast_kernel<<<blocks((long long)INF * HW, TPB), TPB, 0, stream>>>(W_pre, Wpre_t, INF, HW);
    transpose_cast_kernel<<<blocks((long long)INF * HW, TPB), TPB, 0, stream>>>(T_pre, Tpre_t, INF, HW);
    transpose_cast_kernel<<<blocks((long long)INF * HW, TPB), TPB, 0, stream>>>(T_blk, Tblk_t, INF, HW);
    transpose_cast_kernel<<<blocks((long long)INF * OW, TPB), TPB, 0, stream>>>(T_post, Tpost_t, INF, OW);
    transpose_cast_grouped_kernel<<<blocks((long long)WIDTH * HF * HF, TPB), TPB, 0, stream>>>(
        W_blk, Wblk_t, WIDTH, HF, HF);
    transpose_cast_grouped_kernel<<<blocks((long long)WIDTH * HF * OUTF, TPB), TPB, 0, stream>>>(
        W_post, Wpost_t, WIDTH, HF, OUTF);

    // --- stage 1: preprocess ConvSkip ---
    gemm_bf16_kernel<96><<<gemm_grd, TPB, 0, stream>>>(Xbf, Wpre_t, b_pre, G, ntiles);
    hipMemsetAsync(L, 0, (size_t)N * HW * 4, stream);
    laplacian_kernel<96><<<blocks((long long)E * (HW / 4), TPB), TPB, 0, stream>>>(
        G, src, tgt, invdeg, L, E);
    gemm_bf16_kernel<96><<<gemm_grd, TPB, 0, stream>>>(Xbf, Tpre_t, bT_pre, S, ntiles);
    add_relu_bf16_kernel<<<blocks((long long)N * HW, TPB), TPB, 0, stream>>>(L, S, Abf, N * HW);

    // --- stage 2: middle WideConvSkip ---
    grouped_gemm_kernel<HF><<<gemm_grd, TPB, 0, stream>>>(Abf, Wblk_t, b_blk, G, ntiles);
    hipMemsetAsync(L, 0, (size_t)N * HW * 4, stream);
    laplacian_kernel<96><<<blocks((long long)E * (HW / 4), TPB), TPB, 0, stream>>>(
        G, src, tgt, invdeg, L, E);
    gemm_bf16_kernel<96><<<gemm_grd, TPB, 0, stream>>>(Xbf, Tblk_t, bT_blk, S, ntiles);
    add_relu_bf16_kernel<<<blocks((long long)N * HW, TPB), TPB, 0, stream>>>(L, S, Abf, N * HW);

    // --- stage 3: postprocess WideConvSkip ---
    grouped_gemm_kernel<OUTF><<<gemm_grd, TPB, 0, stream>>>(Abf, Wpost_t, b_post, G, ntiles);
    hipMemsetAsync(L, 0, (size_t)N * OW * 4, stream);
    laplacian_kernel<192><<<blocks((long long)E * (OW / 4), TPB), TPB, 0, stream>>>(
        G, src, tgt, invdeg, L, E);
    gemm_bf16_kernel<192><<<gemm_grd, TPB, 0, stream>>>(Xbf, Tpost_t, bT_post, S, ntiles);

    // --- final: relu + width-mean -> [N, 64, 1] ---
    final_mean_kernel<<<blocks((long long)N * OUTF, TPB), TPB, 0, stream>>>(
        L, S, (float*)d_out, N * OUTF);
}
